// Bidirectional_Mamba_72524817760349
// MI455X (gfx1250) — matrix operand order repaired and checked by reading, not compiled, not run on hardware
//
#include <hip/hip_runtime.h>
#include <hip/hip_bf16.h>
#include <cstdint>

// ---- problem constants (from reference) ----
#define NB      8
#define NL      2048
#define NM      (NB * NL)     // 16384 rows
#define DMODEL  256
#define DINNER  512
#define DSTATE  16
#define DTRANK  16
#define NCLS    7

typedef __attribute__((ext_vector_type(16))) __bf16 v16bf;
typedef __attribute__((ext_vector_type(8)))  float  v8f;

__device__ __forceinline__ unsigned short f2bf(float f) {
  unsigned int u = __float_as_uint(f);
  unsigned int r = u + 0x7FFFu + ((u >> 16) & 1u);   // round-to-nearest-even
  return (unsigned short)(r >> 16);
}

union FragAB { v16bf v; unsigned short us[16]; uint4 q[2]; };

__device__ __forceinline__ void load_afrag(FragAB& f, const unsigned short* p) {
  f.q[0] = *(const uint4*)(p);        // elements 0..7  : K = k0+8h .. +7
  f.q[1] = *(const uint4*)(p + 16);   // elements 8..15 : K = k0+16+8h ..
}
__device__ __forceinline__ void load_bfrag(FragAB& f, const unsigned short* p) {
  f.q[0] = *(const uint4*)(p);        // elements 0..7  : K = k0+8h .. +7
  f.q[1] = *(const uint4*)(p + 16);   // elements 8..15 : K = k0+16+8h ..
}

// ---------------------------------------------------------------------------
// bf16 WMMA GEMM:  C[M,N](f32) = A[M,K](bf16) x W[N,K](bf16)^T (+ Cin)
// One wave computes a 32x64 tile: all 6 fragments (2 A sub-tiles + 4 B tiles)
// are preloaded into distinct registers, then 8 independent
// v_wmma_f32_16x16x32_bf16 issue per K-step -> staged loadcnt waits instead
// of a full s_wait_loadcnt 0 before every WMMA, and each B frag is reused 2x.
// ---------------------------------------------------------------------------
__global__ __launch_bounds__(256)
void k_gemm_bf16(const unsigned short* __restrict__ Abuf,
                 const unsigned short* __restrict__ Wbuf,
                 const float* Cin, float* Cout,
                 int M, int N, int K, int acc)
{
  int gtid  = blockIdx.x * 256 + threadIdx.x;
  int wave  = gtid >> 5;
  int lane  = gtid & 31;
  int strips = N >> 6;
  int nwaves = (M >> 5) * strips;
  if (wave >= nwaves) return;
  int mt   = wave / strips;            // 32-row tile index
  int ns   = wave % strips;            // 64-col strip index
  int half = lane >> 4;
  int lrow = lane & 15;

  v8f c[2][4];
  if (acc) {
#pragma unroll
    for (int mi = 0; mi < 2; ++mi)
#pragma unroll
      for (int j = 0; j < 4; ++j)
#pragma unroll
        for (int v = 0; v < 8; ++v)
          c[mi][j][v] = Cin[(size_t)((mt << 5) + (mi << 4) + v + (half << 3)) * N +
                            (ns << 6) + (j << 4) + lrow];
  } else {
#pragma unroll
    for (int mi = 0; mi < 2; ++mi)
#pragma unroll
      for (int j = 0; j < 4; ++j)
#pragma unroll
        for (int v = 0; v < 8; ++v) c[mi][j][v] = 0.f;
  }

  const unsigned short* arow0 = Abuf + (size_t)((mt << 5) + lrow) * K;
  const unsigned short* arow1 = arow0 + (size_t)16 * K;
  const unsigned short* wcol  = Wbuf + (size_t)((ns << 6) + lrow) * K;

  for (int k0 = 0; k0 < K; k0 += 32) {
    FragAB a0, a1, b0, b1, b2, b3;
    load_afrag(a0, arow0 + k0 + (half << 3));
    load_afrag(a1, arow1 + k0 + (half << 3));
    __builtin_prefetch(arow0 + k0 + 32, 0, 3);   // near-cache prefetch of next A chunk
    const unsigned short* wp = wcol + k0 + (half << 3);
    load_bfrag(b0, wp);
    load_bfrag(b1, wp + (size_t)16 * K);
    load_bfrag(b2, wp + (size_t)32 * K);
    load_bfrag(b3, wp + (size_t)48 * K);
    c[0][0] = __builtin_amdgcn_wmma_f32_16x16x32_bf16(false, a0.v, false, b0.v, (short)0, c[0][0], false, false);
    c[1][0] = __builtin_amdgcn_wmma_f32_16x16x32_bf16(false, a1.v, false, b0.v, (short)0, c[1][0], false, false);
    c[0][1] = __builtin_amdgcn_wmma_f32_16x16x32_bf16(false, a0.v, false, b1.v, (short)0, c[0][1], false, false);
    c[1][1] = __builtin_amdgcn_wmma_f32_16x16x32_bf16(false, a1.v, false, b1.v, (short)0, c[1][1], false, false);
    c[0][2] = __builtin_amdgcn_wmma_f32_16x16x32_bf16(false, a0.v, false, b2.v, (short)0, c[0][2], false, false);
    c[1][2] = __builtin_amdgcn_wmma_f32_16x16x32_bf16(false, a1.v, false, b2.v, (short)0, c[1][2], false, false);
    c[0][3] = __builtin_amdgcn_wmma_f32_16x16x32_bf16(false, a0.v, false, b3.v, (short)0, c[0][3], false, false);
    c[1][3] = __builtin_amdgcn_wmma_f32_16x16x32_bf16(false, a1.v, false, b3.v, (short)0, c[1][3], false, false);
  }
#pragma unroll
  for (int mi = 0; mi < 2; ++mi)
#pragma unroll
    for (int j = 0; j < 4; ++j)
#pragma unroll
      for (int v = 0; v < 8; ++v)
        Cout[(size_t)((mt << 5) + (mi << 4) + v + (half << 3)) * N +
             (ns << 6) + (j << 4) + lrow] = c[mi][j][v];
}

// ---------------------------------------------------------------------------
// f32 -> bf16 weight conversion with zero padding (R/C pad for odd shapes)
// ---------------------------------------------------------------------------
__global__ void k_cvt_pad(const float* __restrict__ src, unsigned short* __restrict__ dst,
                          int Rsrc, int Csrc, int Rdst, int Cdst, int total)
{
  int i = blockIdx.x * 256 + threadIdx.x;
  if (i >= total) return;
  int c  = i % Cdst;
  int rb = i / Cdst;
  int r  = rb % Rdst;
  int bt = rb / Rdst;
  float v = (r < Rsrc && c < Csrc) ? src[((size_t)bt * Rsrc + r) * Csrc + c] : 0.f;
  dst[i] = f2bf(v);
}

// ---------------------------------------------------------------------------
// LayerNorm over last dim (256); writes f32 and bf16 (bf16 feeds WMMA GEMM)
// ---------------------------------------------------------------------------
__global__ __launch_bounds__(256)
void k_layernorm(const float* __restrict__ src, const float* __restrict__ w,
                 const float* __restrict__ b, float* __restrict__ dstF,
                 unsigned short* __restrict__ dstB)
{
  __shared__ float red[256];
  int row = blockIdx.x;
  int t   = threadIdx.x;
  float v = src[(size_t)row * DMODEL + t];
  red[t] = v; __syncthreads();
  for (int s = 128; s > 0; s >>= 1) { if (t < s) red[t] += red[t + s]; __syncthreads(); }
  float mu = red[0] * (1.f / DMODEL);
  __syncthreads();
  float d = v - mu;
  red[t] = d * d; __syncthreads();
  for (int s = 128; s > 0; s >>= 1) { if (t < s) red[t] += red[t + s]; __syncthreads(); }
  float var = red[0] * (1.f / DMODEL);
  float o = d * rsqrtf(var + 1e-5f) * w[t] + b[t];
  dstF[(size_t)row * DMODEL + t] = o;
  dstB[(size_t)row * DMODEL + t] = f2bf(o);
}

// ---------------------------------------------------------------------------
// Depthwise causal conv (width 4) + bias + SiLU; dir=+1 taps l-3..l,
// dir=-1 taps l..l+3 (equivalent to conv on flipped sequence, flipped back).
// ---------------------------------------------------------------------------
__global__ __launch_bounds__(256)
void k_conv_silu(const float* __restrict__ xz, const float* __restrict__ cw,
                 const float* __restrict__ cb, float* __restrict__ xc,
                 unsigned short* __restrict__ xcB, int dir)
{
  int idx = blockIdx.x * 256 + threadIdx.x;       // over NM * DINNER
  int d  = idx & (DINNER - 1);
  int bl = idx >> 9;
  int b  = bl >> 11;
  int l  = bl & (NL - 1);
  float s = cb[d];
#pragma unroll
  for (int k = 0; k < 4; ++k) {
    int lt = l + dir * (k - 3);
    if (lt >= 0 && lt < NL)
      s += cw[d * 4 + k] * xz[(size_t)(b * NL + lt) * 1024 + d];  // xi = first half of xz
  }
  float r = s / (1.f + expf(-s));
  xc[idx]  = r;
  xcB[idx] = f2bf(r);
}

// dtr (cols 0..15 of dbl) -> bf16, zero-padded to K=32 for the dt_proj GEMM
__global__ void k_dt_prep(const float* __restrict__ dbl, unsigned short* __restrict__ dtrB)
{
  int i = blockIdx.x * 256 + threadIdx.x;         // over NM * 32
  int c = i & 31, r = i >> 5;
  float v = (c < DTRANK) ? dbl[(size_t)r * 64 + c] : 0.f;
  dtrB[i] = f2bf(v);
}

// dt = softplus(dt_raw + dt_bias) in place
__global__ void k_softplus(float* __restrict__ dt, const float* __restrict__ dtb)
{
  int i = blockIdx.x * 256 + threadIdx.x;         // over NM * DINNER
  int d = i & (DINNER - 1);
  float v = dt[i] + dtb[d];
  dt[i] = (v > 20.f) ? v : log1pf(expf(v));
}

// ---------------------------------------------------------------------------
// Selective-scan. One thread per (b, channel d), 16 fp32 states in registers.
// B/C (shared per (b,l)) double-buffered in LDS; one barrier per timestep.
// dir selects iteration direction; initY=1 seeds y with 2*xc*D (both scans
// of a mixer contribute xc*D once each).
// ---------------------------------------------------------------------------
__global__ __launch_bounds__(256)
void k_scan(const float* __restrict__ xc, const float* __restrict__ dt,
            const float* __restrict__ dbl, const float* __restrict__ Alog,
            const float* __restrict__ Dp, float* __restrict__ y,
            int dir, int initY)
{
  __shared__ float bc[2][32];                     // [0..15]=B, [16..31]=C
  int b = blockIdx.x >> 1;
  int d = ((blockIdx.x & 1) << 8) + threadIdx.x;
  float A[DSTATE], h[DSTATE];
#pragma unroll
  for (int n = 0; n < DSTATE; ++n) { A[n] = -expf(Alog[(size_t)d * DSTATE + n]); h[n] = 0.f; }
  float Dv = Dp[d];
  int l0 = (dir > 0) ? 0 : NL - 1;
  if (threadIdx.x < 32)
    bc[0][threadIdx.x] = dbl[(size_t)(b * NL + l0) * 64 + 16 + threadIdx.x];
  __syncthreads();
  for (int t = 0; t < NL; ++t) {
    int l   = (dir > 0) ? t : (NL - 1 - t);
    int cur = t & 1;
    if (t + 1 < NL && threadIdx.x < 32) {
      int ln = (dir > 0) ? (t + 1) : (NL - 2 - t);
      bc[cur ^ 1][threadIdx.x] = dbl[(size_t)(b * NL + ln) * 64 + 16 + threadIdx.x];
    }
    size_t idx = (size_t)(b * NL + l) * DINNER + d;
    float x = xc[idx], dv = dt[idx];
    float dx = dv * x, acc = 0.f;
#pragma unroll
    for (int n = 0; n < DSTATE; ++n) {
      h[n] = expf(dv * A[n]) * h[n] + dx * bc[cur][n];
      acc += h[n] * bc[cur][16 + n];
    }
    y[idx] = (initY ? (2.f * x * Dv) : y[idx]) + acc;
    __syncthreads();
  }
}

// y *= silu(z); write bf16 for the out_proj WMMA GEMM
__global__ void k_gate(const float* __restrict__ y, const float* __restrict__ xz,
                       unsigned short* __restrict__ yB)
{
  int i  = blockIdx.x * 256 + threadIdx.x;        // over NM * DINNER
  int d  = i & (DINNER - 1);
  int bl = i >> 9;
  float z = xz[(size_t)bl * 1024 + DINNER + d];
  float g = z / (1.f + expf(-z));
  yB[i] = f2bf(y[i] * g);
}

// resid = 2*x  (DEPTH=2 bidirectional residual algebra collapses to this)
__global__ void k_resid_init(const float* __restrict__ x, float* __restrict__ resid)
{
  int i = blockIdx.x * 256 + threadIdx.x;
  resid[i] = 2.f * x[i];
}

// classifier head on the last timestep (tiny: 8x7x256 dots, fp32)
__global__ void k_head(const float* __restrict__ hid, const float* __restrict__ hw,
                       const float* __restrict__ hb, float* __restrict__ out)
{
  int i = threadIdx.x;
  if (i >= NB * NCLS) return;
  int b = i / NCLS, c = i % NCLS;
  const float* hp = hid + (size_t)(b * NL + NL - 1) * DMODEL;
  float s = hb[c];
  for (int k = 0; k < DMODEL; ++k) s += hp[k] * hw[c * DMODEL + k];
  out[i] = s;
}

// ---------------------------------------------------------------------------
extern "C" void kernel_launch(void* const* d_in, const int* in_sizes, int n_in,
                              void* d_out, int out_size, void* d_ws, size_t ws_size,
                              hipStream_t stream)
{
  (void)in_sizes; (void)n_in; (void)out_size; (void)ws_size;
  const float* x     = (const float*)d_in[0];
  const float* inw   = (const float*)d_in[1];
  const float* cw    = (const float*)d_in[2];
  const float* cb    = (const float*)d_in[3];
  const float* xpw   = (const float*)d_in[4];
  const float* dtw   = (const float*)d_in[5];
  const float* dtb   = (const float*)d_in[6];
  const float* Alog  = (const float*)d_in[7];
  const float* Ablog = (const float*)d_in[8];
  const float* Dp    = (const float*)d_in[9];
  const float* outw  = (const float*)d_in[10];
  const float* nw    = (const float*)d_in[11];
  const float* nbv   = (const float*)d_in[12];
  const float* nfw   = (const float*)d_in[13];
  const float* nfb   = (const float*)d_in[14];
  const float* hw    = (const float*)d_in[15];
  const float* hbb   = (const float*)d_in[16];
  float* out = (float*)d_out;

  char* wsp = (char*)d_ws;
  auto alloc = [&](size_t bytes) -> char* {
    char* p = wsp; wsp += (bytes + 255) & ~(size_t)255; return p;
  };
  const size_t M = NM;
  float* xz    = (float*)alloc(M * 1024 * 4);
  float* xc    = (float*)alloc(M * 512 * 4);
  float* dbl   = (float*)alloc(M * 64 * 4);
  float* dt    = (float*)alloc(M * 512 * 4);
  float* y     = (float*)alloc(M * 512 * 4);
  float* resid = (float*)alloc(M * 256 * 4);
  float* lnF   = (float*)alloc(M * 256 * 4);
  unsigned short* lnB   = (unsigned short*)alloc(M * 256 * 2);
  unsigned short* actB  = (unsigned short*)alloc(M * 512 * 2);  // xc_bf16 then y_bf16
  unsigned short* dtrB  = (unsigned short*)alloc(M * 32 * 2);
  unsigned short* winB  = (unsigned short*)alloc((size_t)2 * 1024 * 256 * 2);
  unsigned short* wxpB  = (unsigned short*)alloc((size_t)2 * 64 * 512 * 2);   // pad N 48->64
  unsigned short* wdtB  = (unsigned short*)alloc((size_t)2 * 512 * 32 * 2);   // pad K 16->32
  unsigned short* woutB = (unsigned short*)alloc((size_t)2 * 256 * 512 * 2);

  auto cvt = [&](const float* s, unsigned short* dd, int Rs, int Cs, int Rd, int Cd, int batch) {
    int total = batch * Rd * Cd;
    k_cvt_pad<<<(total + 255) / 256, 256, 0, stream>>>(s, dd, Rs, Cs, Rd, Cd, total);
  };
  cvt(inw,  winB,  1024, 256, 1024, 256, 2);
  cvt(xpw,  wxpB,  48,   512, 64,   512, 2);
  cvt(dtw,  wdtB,  512,  16,  512,  32,  2);
  cvt(outw, woutB, 256,  512, 256,  512, 2);

  auto gemm = [&](const unsigned short* A, const unsigned short* W,
                  const float* Cin, float* Cout, int Mm, int Nn, int Kk, int acc) {
    int waves  = (Mm >> 5) * (Nn >> 6);            // 32x64 tile per wave
    int blocks = (waves * 32 + 255) / 256;
    k_gemm_bf16<<<blocks, 256, 0, stream>>>(A, W, Cin, Cout, Mm, Nn, Kk, acc);
  };

  k_resid_init<<<(int)(M * 256 / 256), 256, 0, stream>>>(x, resid);

  for (int layer = 0; layer < 2; ++layer) {
    int dir = (layer == 0) ? 1 : -1;
    // hn = LN(x, norm_w[layer], norm_b[layer])  (flip-invariant; dir handled below)
    k_layernorm<<<(int)M, 256, 0, stream>>>(x, nw + layer * 256, nbv + layer * 256, lnF, lnB);
    // xz = hn @ in_proj_w^T         [M x 1024, K=256]  -> WMMA
    gemm(lnB, winB + (size_t)layer * 1024 * 256, nullptr, xz, (int)M, 1024, 256, 0);
    // xc = silu(conv1d(xi) + b)     (direction-aware taps)
    k_conv_silu<<<(int)(M * 512 / 256), 256, 0, stream>>>(
        xz, cw + layer * DINNER * 4, cb + layer * DINNER, xc, actB, dir);
    // dbl = xc @ x_proj_w^T         [M x 64(pad), K=512] -> WMMA
    gemm(actB, wxpB + (size_t)layer * 64 * 512, nullptr, dbl, (int)M, 64, 512, 0);
    k_dt_prep<<<(int)(M * 32 / 256), 256, 0, stream>>>(dbl, dtrB);
    // dt_raw = dtr @ dt_proj_w^T    [M x 512, K=32(pad)] -> WMMA
    gemm(dtrB, wdtB + (size_t)layer * 512 * 32, nullptr, dt, (int)M, 512, 32, 0);
    k_softplus<<<(int)(M * 512 / 256), 256, 0, stream>>>(dt, dtb + layer * DINNER);
    // y = scan_dir(A) + scan_{-dir}(Ab) + 2*xc*D
    k_scan<<<16, 256, 0, stream>>>(xc, dt, dbl, Alog + (size_t)layer * DINNER * DSTATE,
                                   Dp + layer * DINNER, y, dir, 1);
    k_scan<<<16, 256, 0, stream>>>(xc, dt, dbl, Ablog + (size_t)layer * DINNER * DSTATE,
                                   Dp + layer * DINNER, y, -dir, 0);
    // y *= silu(z)
    k_gate<<<(int)(M * 512 / 256), 256, 0, stream>>>(y, xz, actB);
    // resid += y @ out_proj_w^T     [M x 256, K=512] -> WMMA with C accumulate
    gemm(actB, woutB + (size_t)layer * 256 * 512, resid, resid, (int)M, 256, 512, 1);
  }

  // final LN + head on last timestep
  k_layernorm<<<(int)M, 256, 0, stream>>>(resid, nfw, nfb, lnF, lnB);
  k_head<<<1, 64, 0, stream>>>(lnF, hw, hbb, out);
}
